// SelfAttnModel_55551107006662
// MI455X (gfx1250) — compile-verified, matrix operand order checked by reading, not run on hardware
//
#include <hip/hip_runtime.h>
#include <hip/hip_bf16.h>

#define BSZ   4
#define HWSZ  4096
#define CDIM  256
#define DQK   16

typedef __attribute__((ext_vector_type(2)))  float  v2f;
typedef __attribute__((ext_vector_type(8)))  float  v8f;
typedef __attribute__((ext_vector_type(16))) __bf16 v16bf;

// 16-bit WMMA operand K-index mapping: element j of lane (half = lane>>4):
//   k = (j&7) + 8*half + 16*(j>>3)      (5-bit decomposition: [2:0],[3],[4])
__device__ __forceinline__ int kmap16(int j, int half) {
    return (j & 7) + 8 * half + 16 * (j >> 3);
}

// ---------------------------------------------------------------------------
// Kernel 0: repack weights into WMMA-B-fragment order (bf16).
//   wv_p: [kc(8)][col(256)][half(2)][j(16)]  (128 KB)
//   wq_p/wk_p: [kc(8)][col(16)][half(2)][j(16)]  (8 KB each)
// value at [kc][col][h][j] = w[(kc*32 + kmap16(j,h)) * ncols + col]
// ---------------------------------------------------------------------------
__global__ __launch_bounds__(256) void pack_weights(
    const float* __restrict__ wq, const float* __restrict__ wk,
    const float* __restrict__ wv,
    __bf16* __restrict__ wq_p, __bf16* __restrict__ wk_p,
    __bf16* __restrict__ wv_p)
{
    const int idx = blockIdx.x * 256 + threadIdx.x;   // [0, 65536)
    {
        const int kc  = idx >> 13;
        const int col = (idx >> 5) & 255;
        const int t   = idx & 31;
        const int k   = kc * 32 + kmap16(t & 15, t >> 4);
        wv_p[idx] = (__bf16)wv[(size_t)k * CDIM + col];
    }
    if (idx < 8 * DQK * 32) {                          // [0, 4096)
        const int kc  = idx >> 9;
        const int col = (idx >> 5) & 15;
        const int t   = idx & 31;
        const int k   = kc * 32 + kmap16(t & 15, t >> 4);
        wq_p[idx] = (__bf16)wq[(size_t)k * DQK + col];
        wk_p[idx] = (__bf16)wk[(size_t)k * DQK + col];
    }
}

// ---------------------------------------------------------------------------
// Kernel 1: 1x1-conv projections.  x as [B*HW, C] row-major, 16 rows per WG.
//  - x tile staged to LDS with global_load_async_to_lds_b128 (ASYNCcnt path)
//  - q/k stored f32 with lane-permuted rows for the f32-WMMA B/A reads:
//      pos(d) = ((d&3)>>1)*8 + (d>>2)*2 + (d&1)
//  - V stored bf16 in packed B-fragment order per batch:
//      [kc(128)][col(256)][half(2)][j(16)]
// ---------------------------------------------------------------------------
__global__ __launch_bounds__(256) void proj_kernel(
    const float* __restrict__ x,
    const float* __restrict__ bq, const float* __restrict__ bk,
    const float* __restrict__ bv,
    const __bf16* __restrict__ wq_p, const __bf16* __restrict__ wk_p,
    const __bf16* __restrict__ wv_p,
    float* __restrict__ q_ws, float* __restrict__ k_ws,
    __bf16* __restrict__ v_ws)
{
    __shared__ float xt[16][CDIM];      // 16 KB f32 x tile

    const int row0 = blockIdx.x * 16;   // flattened row in [0, B*HW)
    const int tid  = threadIdx.x;

    // ---- async-copy the contiguous 16KB tile into LDS (4 x b128 / thread) ----
    {
        const unsigned int lds0 = (unsigned int)(size_t)&xt[0][0];
        const float* gbase = x + (size_t)row0 * CDIM;
        #pragma unroll
        for (int q2 = 0; q2 < 4; ++q2) {
            const int e0 = (q2 * 256 + tid) * 4;      // float index in tile
            asm volatile("global_load_async_to_lds_b128 %0, %1, off"
                         :: "v"(lds0 + e0 * 4),
                            "v"((unsigned long long)(size_t)(gbase + e0))
                         : "memory");
        }
        asm volatile("s_wait_asynccnt 0x0" ::: "memory");
    }
    __syncthreads();

    const int lane = tid & 31;
    const int wave = tid >> 5;
    const int n    = lane & 15;    // A-matrix M row / B-matrix N col
    const int half = lane >> 4;

    // A fragments for all 8 K-chunks of 32 (K = 256 total)
    v16bf a[8];
    #pragma unroll
    for (int kc = 0; kc < 8; ++kc) {
        #pragma unroll
        for (int j = 0; j < 16; ++j)
            a[kc][j] = (__bf16)xt[n][kc * 32 + kmap16(j, half)];
    }

    // --- V projection: wave owns output cols [wave*32, wave*32+32) ---
    #pragma unroll
    for (int t = 0; t < 2; ++t) {
        const int col0 = wave * 32 + t * 16;
        v8f acc = {};
        #pragma unroll
        for (int kc = 0; kc < 8; ++kc) {
            const v16bf bfrag = *(const v16bf*)
                (wv_p + ((size_t)(kc * CDIM + col0 + n) * 32) + half * 16);
            acc = __builtin_amdgcn_wmma_f32_16x16x32_bf16(
                      false, a[kc], false, bfrag, (short)0, acc, false, false);
        }
        const float bias = bv[col0 + n];
        #pragma unroll
        for (int r = 0; r < 8; ++r) {
            const int mm  = r + 8 * half;        // C layout: M = r + 8*half
            const int kg  = row0 + mm;           // global key index
            const int bb  = kg >> 12;            // batch
            const int key = kg & (HWSZ - 1);
            const int kc  = key >> 5, o = key & 31;
            const int j   = (o & 7) + 8 * (o >> 4);
            const int hB  = (o >> 3) & 1;
            v_ws[(size_t)bb * HWSZ * CDIM +
                 (size_t)(kc * CDIM + col0 + n) * 32 + hB * 16 + j] =
                (__bf16)(acc[r] + bias);
        }
    }

    // --- Q / K projections: waves 0 and 1 ---
    if (wave < 2) {
        const __bf16* wp   = (wave == 0) ? wq_p : wk_p;
        const float*  bvec = (wave == 0) ? bq : bk;
        float*        op   = (wave == 0) ? q_ws : k_ws;
        v8f acc = {};
        #pragma unroll
        for (int kc = 0; kc < 8; ++kc) {
            const v16bf bfrag = *(const v16bf*)
                (wp + ((size_t)(kc * DQK + n) * 32) + half * 16);
            acc = __builtin_amdgcn_wmma_f32_16x16x32_bf16(
                      false, a[kc], false, bfrag, (short)0, acc, false, false);
        }
        const float bias = bvec[n];
        // permuted store so energy-WMMA operands read 8 contiguous floats
        const int pos = ((n & 3) >> 1) * 8 + (n >> 2) * 2 + (n & 1);
        #pragma unroll
        for (int r = 0; r < 8; ++r) {
            const int mm = r + 8 * half;
            op[(size_t)(row0 + mm) * DQK + pos] = acc[r] + bias;
        }
    }
}

// ---------------------------------------------------------------------------
// Kernel 2: fused energy -> softmax -> attention write -> A@V -> gamma*out+x.
// One WG per (batch, 16 query rows).  8 waves, 256 threads.
// ---------------------------------------------------------------------------
__global__ __launch_bounds__(256) void attn_kernel(
    const float* __restrict__ x,
    const float* __restrict__ q_ws, const float* __restrict__ k_ws,
    const __bf16* __restrict__ v_ws, const float* __restrict__ gamma_p,
    float* __restrict__ out, float* __restrict__ attn)
{
    __shared__ __bf16 p_pack[16][4][2][16];  // 4 KB: p tile, fragment order
    __shared__ float  red[8][16];
    __shared__ float  rowinv[16];

    const int b    = blockIdx.y;
    const int m0   = blockIdx.x * 16;
    const int tid  = threadIdx.x;
    const int lane = tid & 31;
    const int wave = tid >> 5;
    const int n    = lane & 15;
    const int half = lane >> 4;

    const float*  qb    = q_ws + (size_t)b * HWSZ * DQK;
    const float*  kb    = k_ws + (size_t)b * HWSZ * DQK;
    const __bf16* vb    = v_ws + (size_t)b * HWSZ * CDIM;
    float*        attnb = attn + (size_t)b * HWSZ * HWSZ + (size_t)m0 * HWSZ;

    // Q fragments: permuted row -> 8 contiguous floats per lane
    v2f qa[4];
    {
        const float* qr = qb + (size_t)(m0 + n) * DQK + half * 8;
        #pragma unroll
        for (int i = 0; i < 4; ++i) { qa[i][0] = qr[2*i]; qa[i][1] = qr[2*i+1]; }
    }

    // energy tile (fp32 WMMA chain, K=16) for 16 keys at key0
    auto energy_tile = [&](int key0) -> v8f {
        const float* kr = kb + (size_t)(key0 + n) * DQK + half * 8;
        float k8[8];
        #pragma unroll
        for (int p = 0; p < 8; ++p) k8[p] = kr[p];
        v8f acc = {};
        #pragma unroll
        for (int i = 0; i < 4; ++i) {
            v2f bfrag; bfrag[0] = k8[2*i]; bfrag[1] = k8[2*i+1];
            acc = __builtin_amdgcn_wmma_f32_16x16x4_f32(
                      false, qa[i], false, bfrag, (short)0, acc, false, false);
        }
        return acc;
    };

    // ---------------- Sweep 1: rowsum of exp(energy) ----------------
    float psum[8];
    #pragma unroll
    for (int r = 0; r < 8; ++r) psum[r] = 0.0f;

    for (int ch = wave; ch < HWSZ / 16; ch += 8) {
        const int key0 = ch * 16;
        if (ch + 8 < HWSZ / 16)   // global_prefetch_b8 of next K chunk
            __builtin_prefetch(kb + (size_t)(key0 + 128 + n) * DQK, 0, 0);
        v8f e = energy_tile(key0);
        #pragma unroll
        for (int r = 0; r < 8; ++r) psum[r] += __expf(e[r]);
    }
    #pragma unroll
    for (int r = 0; r < 8; ++r) {
        float s = psum[r];
        s += __shfl_xor(s, 1, 32);
        s += __shfl_xor(s, 2, 32);
        s += __shfl_xor(s, 4, 32);
        s += __shfl_xor(s, 8, 32);
        psum[r] = s;
    }
    if (n == 0) {
        #pragma unroll
        for (int r = 0; r < 8; ++r) red[wave][r + 8 * half] = psum[r];
    }
    __syncthreads();
    if (tid < 16) {
        float s = 0.0f;
        #pragma unroll
        for (int w = 0; w < 8; ++w) s += red[w][tid];
        rowinv[tid] = 1.0f / s;
    }
    __syncthreads();
    float inv[8];
    #pragma unroll
    for (int r = 0; r < 8; ++r) inv[r] = rowinv[r + 8 * half];

    // ------- Sweep 2: normalize, write attention, A@V (bf16 WMMA) -------
    const float g = gamma_p[0];
    v8f o0 = {}, o1 = {};

    // writer-side permutation constants for this wave's 16 keys in the block
    const int kc_w = wave >> 1;
    const int o_w  = ((wave & 1) << 4) + n;
    const int j_w  = (o_w & 7) + 8 * (o_w >> 4);
    const int h_w  = (o_w >> 3) & 1;

    for (int big = 0; big < HWSZ / 128; ++big) {
        const int key0 = big * 128 + wave * 16;   // this wave's 16 keys
        v8f e = energy_tile(key0);
        #pragma unroll
        for (int r = 0; r < 8; ++r) {
            const int mm = r + 8 * half;
            const float p = __expf(e[r]) * inv[r];
            attnb[(size_t)mm * HWSZ + key0 + n] = p;   // attention output
            p_pack[mm][kc_w][h_w][j_w] = (__bf16)p;    // fragment-order stage
        }
        __syncthreads();

        // A fragments: contiguous 32B LDS reads (2 x ds_load_b128 each)
        v16bf pa[4];
        #pragma unroll
        for (int kc = 0; kc < 4; ++kc)
            pa[kc] = *(const v16bf*)&p_pack[n][kc][half][0];

        #pragma unroll
        for (int t = 0; t < 2; ++t) {
            const int col0 = wave * 32 + t * 16;
            v8f o = t ? o1 : o0;
            #pragma unroll
            for (int kc = 0; kc < 4; ++kc) {
                const v16bf bfrag = *(const v16bf*)
                    (vb + (size_t)((big * 4 + kc) * CDIM + col0 + n) * 32
                        + half * 16);
                o = __builtin_amdgcn_wmma_f32_16x16x32_bf16(
                        false, pa[kc], false, bfrag, (short)0, o, false, false);
            }
            if (t) o1 = o; else o0 = o;
        }
        __syncthreads();   // protect p_pack before next block overwrites
    }

    // ---------------- Epilogue: out = o*gamma + x ----------------
    const float* xb   = x   + ((size_t)b * HWSZ + m0) * CDIM;
    float*       outb = out + ((size_t)b * HWSZ + m0) * CDIM;
    #pragma unroll
    for (int t = 0; t < 2; ++t) {
        const int col0 = wave * 32 + t * 16;
        v8f o = t ? o1 : o0;
        #pragma unroll
        for (int r = 0; r < 8; ++r) {
            const int mm = r + 8 * half;
            outb[(size_t)mm * CDIM + col0 + n] =
                o[r] * g + xb[(size_t)mm * CDIM + col0 + n];
        }
    }
}

// ---------------------------------------------------------------------------
extern "C" void kernel_launch(void* const* d_in, const int* in_sizes, int n_in,
                              void* d_out, int out_size, void* d_ws, size_t ws_size,
                              hipStream_t stream) {
    const float* x     = (const float*)d_in[0];
    const float* wq    = (const float*)d_in[1];
    const float* bq    = (const float*)d_in[2];
    const float* wk    = (const float*)d_in[3];
    const float* bk    = (const float*)d_in[4];
    const float* wv    = (const float*)d_in[5];
    const float* bv    = (const float*)d_in[6];
    const float* gamma = (const float*)d_in[7];

    float* out  = (float*)d_out;                               // [B,H,W,C]
    float* attn = out + (size_t)BSZ * HWSZ * CDIM;             // [B,HW,HW]

    // workspace: q f32 | k f32 | v bf16 packed | packed weights
    float*  q_ws = (float*)d_ws;
    float*  k_ws = q_ws + (size_t)BSZ * HWSZ * DQK;
    __bf16* v_ws = (__bf16*)(k_ws + (size_t)BSZ * HWSZ * DQK);
    __bf16* wv_p = v_ws + (size_t)BSZ * HWSZ * CDIM;
    __bf16* wq_p = wv_p + (size_t)8 * CDIM * 32;
    __bf16* wk_p = wq_p + (size_t)8 * DQK * 32;

    pack_weights<<<256, 256, 0, stream>>>(wq, wk, wv, wq_p, wk_p, wv_p);

    proj_kernel<<<(BSZ * HWSZ) / 16, 256, 0, stream>>>(
        x, bq, bk, bv, wq_p, wk_p, wv_p, q_ws, k_ws, v_ws);

    attn_kernel<<<dim3(HWSZ / 16, BSZ), 256, 0, stream>>>(
        x, q_ws, k_ws, v_ws, gamma, out, attn);
}
